// SuperGlueAttentionalPropagation_29927332119005
// MI455X (gfx1250) — matrix operand order repaired and checked by reading, not compiled, not run on hardware
//
#include <hip/hip_runtime.h>
#include <cstddef>
#include <cstdint>

// SuperGlue attentional propagation, CDNA5/gfx1250 (wave32, WMMA).
// B=8, D=256, N=2048, H=4, HD=64.
//
// All matmuls via V_WMMA_F32_16X16X32_F16.  Every f16 tensor (activations
// AND weights) is stored pre-swizzled in its consumer's WMMA fragment layout
// so fragment loads are contiguous 32 B per lane (2 x b128).
//
// Attention: K/V blocks staged to LDS with GLOBAL_LOAD_ASYNC_TO_LDS_B128
// (ASYNCcnt), double-buffered, shared by 8 waves; softmax row sums computed
// with an extra WMMA against an all-ones fragment.
//
// Workspace: 6 x 8.39 MB activation slots + 1.31 MB packed weights = 51.7 MB.

typedef __attribute__((ext_vector_type(16))) _Float16 v16h;
typedef __attribute__((ext_vector_type(8)))  float    v8f;

#define BN_EPS 1e-5f

__device__ __forceinline__ v8f wmma16(v16h a, v16h b, v8f c) {
  return __builtin_amdgcn_wmma_f32_16x16x32_f16(false, a, false, b, (short)0, c,
                                                false, false);
}

// Async global->LDS copy, 16 B per lane.  Tracked with ASYNCcnt.
__device__ __forceinline__ void async_b128(uint32_t lds_byte,
                                           const _Float16* g) {
  asm volatile("global_load_async_to_lds_b128 %0, %1, off"
               :: "v"(lds_byte), "v"(g)
               : "memory");
}
__device__ __forceinline__ void wait_async0() {
  asm volatile("s_wait_asynccnt 0x0" ::: "memory");
}

// Inverse fragment maps (k -> (hi,e)) for swizzled stores.
__device__ __forceinline__ void a_inv(int k, int& hi, int& e) {
  hi = k >> 4;
  e = k & 15;
}
__device__ __forceinline__ void b_inv(int k, int& hi, int& e) {
  hi = k >> 4;
  e = k & 15;
}

// ---------------------------------------------------------------------------
// Pack f32 activations [8,256,2048] -> f16 B-fragment swizzle (KS = 8).
// ---------------------------------------------------------------------------
__global__ __launch_bounds__(256) void pack_kernel(const float* __restrict__ X,
                                                   _Float16* __restrict__ out) {
  const int tid = blockIdx.x * 256 + threadIdx.x;  // 262144 total
  const int q = tid & 127;
  const int i = (tid >> 7) & 255;
  const int b = tid >> 15;
  const float* src = X + ((size_t)b * 256 + i) * 2048 + q * 16;
  float v[16];
  *(float4*)(v + 0)  = *(const float4*)(src + 0);
  *(float4*)(v + 4)  = *(const float4*)(src + 4);
  *(float4*)(v + 8)  = *(const float4*)(src + 8);
  *(float4*)(v + 12) = *(const float4*)(src + 12);
  const int coltile = b * 128 + q;
  const int kstep = i >> 5;
  int hic, e;
  b_inv(i & 31, hic, e);
  _Float16* dst = out + (size_t)(coltile * 8 + kstep) * 512 + e;
#pragma unroll
  for (int j = 0; j < 16; ++j)
    dst[(size_t)(j + 16 * hic) * 16] = (_Float16)v[j];
}

// ---------------------------------------------------------------------------
// Pack f32 weights [O, ICH] -> f16 A-fragment swizzle.
// One thread per (o, 32-wide kstep); O*(ICH/32) threads.
// ---------------------------------------------------------------------------
__global__ __launch_bounds__(256) void packw_kernel(const float* __restrict__ W,
                                                    _Float16* __restrict__ out,
                                                    int O, int ICH) {
  const int ks_total = ICH >> 5;
  const int tid = blockIdx.x * 256 + threadIdx.x;
  if (tid >= O * ks_total) return;
  const int kstep = tid % ks_total;
  const int o = tid / ks_total;
  const float* src = W + (size_t)o * ICH + kstep * 32;
  float v[32];
#pragma unroll
  for (int j = 0; j < 8; ++j)
    *(float4*)(v + 4 * j) = *(const float4*)(src + 4 * j);
  const int rowtile = o >> 4;
  const int ln = o & 15;
  _Float16* dst = out + (size_t)((rowtile * ks_total + kstep) * 32) * 16;
#pragma unroll
  for (int k = 0; k < 32; ++k) {
    int hi, e;
    a_inv(k, hi, e);
    dst[(ln + 16 * hi) * 16 + e] = (_Float16)v[k];
  }
}

// ---------------------------------------------------------------------------
// Generic GEMM:  C[O, 16384] = Wsw @ X + bias (+ epilogue).  All-f16 operands.
// Wave tile 32x64; block = 4 waves -> 32x256.  Grid: (64, O/32).
// ---------------------------------------------------------------------------
enum { M_PROJ_QA = 0, M_PROJ_KB = 1, M_PROJ_VA = 2, M_WM = 3, M_MLP1 = 4,
       M_MLP2 = 5 };

template <int MODE, int ICH>
__global__ __launch_bounds__(128) void gemm_kernel(
    const _Float16* __restrict__ Wsw, const float* __restrict__ bias,
    const _Float16* __restrict__ Xa, const _Float16* __restrict__ Xb,
    _Float16* __restrict__ outH, float* __restrict__ outF,
    const float* __restrict__ bn_g, const float* __restrict__ bn_b,
    const float* __restrict__ bn_m, const float* __restrict__ bn_v) {
  constexpr int KS_IN = (MODE == M_MLP2) ? 16 : 8;
  constexpr int KSW = ICH >> 5;
  const int lane = threadIdx.x & 31;
  const int wave = threadIdx.x >> 5;
  const int hi = lane >> 4;
  const int ln = lane & 15;
  const int row0 = blockIdx.y * 32;
  const int rt0 = blockIdx.y * 2;
  const int col0 = blockIdx.x * 256 + wave * 64;
  const int ct0 = col0 >> 4;

  v8f acc[2][4] = {};

  for (int k0 = 0; k0 < ICH; k0 += 32) {
    const int ksw = k0 >> 5;
    v16h a[2];
#pragma unroll
    for (int t = 0; t < 2; ++t)
      a[t] = *(const v16h*)(Wsw +
               ((size_t)((rt0 + t) * KSW + ksw) * 32 + lane) * 16);
    const _Float16* srcp;
    int ks;
    if constexpr (MODE == M_MLP1) {
      if (k0 < 256) { srcp = Xa; ks = k0 >> 5; }
      else          { srcp = Xb; ks = (k0 - 256) >> 5; }
    } else {
      srcp = Xa; ks = k0 >> 5;
    }
    v16h bf[4];
#pragma unroll
    for (int c = 0; c < 4; ++c)
      bf[c] = *(const v16h*)(srcp +
                ((size_t)((ct0 + c) * KS_IN + ks) * 32 + lane) * 16);
#pragma unroll
    for (int t = 0; t < 2; ++t)
#pragma unroll
      for (int c = 0; c < 4; ++c) acc[t][c] = wmma16(a[t], bf[c], acc[t][c]);
  }

#pragma unroll
  for (int t = 0; t < 2; ++t) {
#pragma unroll
    for (int r = 0; r < 8; ++r) {
      const int o = row0 + t * 16 + r + hi * 8;
      const float bs = bias[o];
      float sc = 0.f, sh = 0.f;
      if constexpr (MODE == M_MLP1) {
        sc = bn_g[o] * rsqrtf(bn_v[o] + BN_EPS);
        sh = bn_b[o] - bn_m[o] * sc;
      }
#pragma unroll
      for (int c = 0; c < 4; ++c) {
        const int col = col0 + c * 16 + ln;
        const int b = col >> 11;
        const int n = col & 2047;
        float v = acc[t][c][r] + bs;
        if constexpr (MODE == M_PROJ_QA) {
          const int d = o >> 2, h = o & 3;
          const size_t head = (size_t)(b * 4 + h) * 131072;
          int hq, e; a_inv(d & 31, hq, e);
          outH[head + ((size_t)((n >> 4) * 2 + (d >> 5)) * 32 +
                       (ln + 16 * hq)) * 16 + e] = (_Float16)v;
        } else if constexpr (MODE == M_PROJ_KB) {
          const int d = o >> 2, h = o & 3;
          const size_t head = (size_t)(b * 4 + h) * 131072;
          int hk, e; b_inv(d & 31, hk, e);
          outH[head + ((size_t)((n >> 4) * 2 + (d >> 5)) * 32 +
                       (ln + 16 * hk)) * 16 + e] = (_Float16)v;
        } else if constexpr (MODE == M_PROJ_VA) {
          const int d = o >> 2, h = o & 3;
          const size_t head = (size_t)(b * 4 + h) * 131072;
          int hv, e; a_inv(n & 31, hv, e);
          outH[head + ((size_t)((n >> 5) * 4 + (d >> 4)) * 32 +
                       ((d & 15) + 16 * hv)) * 16 + e] = (_Float16)v;
        } else if constexpr (MODE == M_WM || MODE == M_MLP1) {
          constexpr int KS_OUT = (MODE == M_MLP1) ? 16 : 8;
          if constexpr (MODE == M_MLP1) v = fmaxf(v * sc + sh, 0.f);
          int ho, e; b_inv(o & 31, ho, e);
          outH[((size_t)((col >> 4) * KS_OUT + (o >> 5)) * 32 +
                (ln + 16 * ho)) * 16 + e] = (_Float16)v;
        } else {
          outF[((size_t)b * 256 + o) * 2048 + n] = v;
        }
      }
    }
  }
}

// ---------------------------------------------------------------------------
// Flash attention.  Block = 256 threads = 8 waves; each wave owns 16 query
// rows.  K/V blocks (32 keys x 64 d, 4 KB each) async-DMA'd into LDS,
// double-buffered, shared by all 8 waves.  Grid = B*H*(N/128) = 512.
// ---------------------------------------------------------------------------
__global__ __launch_bounds__(256) void attn_kernel(
    const _Float16* __restrict__ Q, const _Float16* __restrict__ K,
    const _Float16* __restrict__ V, _Float16* __restrict__ O) {
  __shared__ alignas(32) _Float16 sK[2][2048];   // 2 x 4 KB
  __shared__ alignas(32) _Float16 sV[2][2048];   // 2 x 4 KB
  __shared__ alignas(32) _Float16 sPb[8][512];   // per-wave P (B-frag order)
  __shared__ float sScale[8][16], sTs[8][16], sInv[8][16];

  const int tid = threadIdx.x;
  const int lane = tid & 31;
  const int wave = tid >> 5;
  const int hi = lane >> 4;
  const int ln = lane & 15;

  const int blk = blockIdx.x;
  const int nblk = blk & 15;
  const int h = (blk >> 4) & 3;
  const int b = blk >> 6;
  const int n0 = nblk * 128 + wave * 16;
  const int ntile = n0 >> 4;

  const _Float16* Qh = Q + (size_t)(b * 4 + h) * 131072;
  const _Float16* Kh = K + (size_t)(b * 4 + h) * 131072;
  const _Float16* Vh = V + (size_t)(b * 4 + h) * 131072;

  const uint32_t ldsK[2] = {(uint32_t)(uintptr_t)&sK[0][0],
                            (uint32_t)(uintptr_t)&sK[1][0]};
  const uint32_t ldsV[2] = {(uint32_t)(uintptr_t)&sV[0][0],
                            (uint32_t)(uintptr_t)&sV[1][0]};

  v16h qa0 = *(const v16h*)(Qh + ((size_t)(ntile * 2 + 0) * 32 + lane) * 16);
  v16h qa1 = *(const v16h*)(Qh + ((size_t)(ntile * 2 + 1) * 32 + lane) * 16);

  v16h ones;
#pragma unroll
  for (int e = 0; e < 16; ++e) ones[e] = (_Float16)1.0f;

  float runmax[8], runsum[8];
#pragma unroll
  for (int r = 0; r < 8; ++r) { runmax[r] = -1e30f; runsum[r] = 0.f; }
  v8f oacc[4] = {};

  auto issue = [&](int m, int buf) {
    async_b128(ldsK[buf] + tid * 16, Kh + (size_t)(m >> 4) * 1024 + tid * 8);
    async_b128(ldsV[buf] + tid * 16, Vh + (size_t)(m >> 5) * 2048 + tid * 8);
  };
  issue(0, 0);

  for (int m0 = 0; m0 < 2048; m0 += 32) {
    const int buf = (m0 >> 5) & 1;
    wait_async0();
    __syncthreads();
    if (m0 + 32 < 2048) issue(m0 + 32, buf ^ 1);

    const _Float16* kb_base = &sK[buf][0];
    const _Float16* vb_base = &sV[buf][0];

    v8f s[2];
#pragma unroll
    for (int ms = 0; ms < 2; ++ms) {
      v16h kb0 = *(const v16h*)(kb_base + (ms * 2 + 0) * 512 + lane * 16);
      v16h kb1 = *(const v16h*)(kb_base + (ms * 2 + 1) * 512 + lane * 16);
      v8f z = {};
      z = wmma16(qa0, kb0, z);
      z = wmma16(qa1, kb1, z);
      s[ms] = z;
    }

    float scr[8];
#pragma unroll
    for (int r = 0; r < 8; ++r) {
      const float v0 = s[0][r] * 0.125f;
      const float v1 = s[1][r] * 0.125f;
      float mx = fmaxf(v0, v1);
#pragma unroll
      for (int off = 1; off < 16; off <<= 1)
        mx = fmaxf(mx, __shfl_xor(mx, off, 32));
      const float newm = fmaxf(runmax[r], mx);
      scr[r] = __expf(runmax[r] - newm);
      runmax[r] = newm;
      const float p0 = __expf(v0 - newm);
      const float p1 = __expf(v1 - newm);
      sPb[wave][(r + hi * 8) * 16 + ln] = (_Float16)p0;
      sPb[wave][(r + hi * 8 + 16) * 16 + ln] = (_Float16)p1;
      if (ln == 0) sScale[wave][r + hi * 8] = scr[r];
    }

    const v16h pb = *(const v16h*)(&sPb[wave][lane * 16]);

    v8f rs = {};
    rs = wmma16(ones, pb, rs);
    if (hi == 0) sTs[wave][ln] = rs[0];

    const float fsc = sScale[wave][ln];
#pragma unroll
    for (int t = 0; t < 4; ++t)
#pragma unroll
      for (int r = 0; r < 8; ++r) oacc[t][r] *= fsc;

#pragma unroll
    for (int t = 0; t < 4; ++t) {
      v16h va = *(const v16h*)(vb_base + t * 512 + lane * 16);
      oacc[t] = wmma16(va, pb, oacc[t]);
    }

#pragma unroll
    for (int r = 0; r < 8; ++r)
      runsum[r] = runsum[r] * scr[r] + sTs[wave][r + hi * 8];
  }

  if (ln == 0) {
#pragma unroll
    for (int r = 0; r < 8; ++r) sInv[wave][r + hi * 8] = runsum[r];
  }
  const float inv = 1.f / sInv[wave][ln];
  const int coltile = (b * 2048 + n0) >> 4;
#pragma unroll
  for (int t = 0; t < 4; ++t)
#pragma unroll
    for (int r = 0; r < 8; ++r) {
      const int d = t * 16 + r + hi * 8;
      const int ic = 4 * d + h;
      int ho, e;
      b_inv(ic & 31, ho, e);
      O[((size_t)(coltile * 8 + (ic >> 5)) * 32 + (ln + 16 * ho)) * 16 + e] =
          (_Float16)(oacc[t][r] * inv);
    }
}

// ---------------------------------------------------------------------------
extern "C" void kernel_launch(void* const* d_in, const int* in_sizes, int n_in,
                              void* d_out, int out_size, void* d_ws,
                              size_t ws_size, hipStream_t stream) {
  const float* x   = (const float*)d_in[0];
  const float* src = (const float*)d_in[1];
  const float* Wq = (const float*)d_in[2];  const float* bq = (const float*)d_in[3];
  const float* Wk = (const float*)d_in[4];  const float* bk = (const float*)d_in[5];
  const float* Wv = (const float*)d_in[6];  const float* bv = (const float*)d_in[7];
  const float* Wm = (const float*)d_in[8];  const float* bm = (const float*)d_in[9];
  const float* W1 = (const float*)d_in[10]; const float* b1 = (const float*)d_in[11];
  const float* g  = (const float*)d_in[12]; const float* be = (const float*)d_in[13];
  const float* mu = (const float*)d_in[14]; const float* vr = (const float*)d_in[15];
  const float* W2 = (const float*)d_in[16]; const float* b2 = (const float*)d_in[17];
  float* out = (float*)d_out;

  char* ws = (char*)d_ws;
  const size_t SZ = (size_t)8 * 256 * 2048 * sizeof(_Float16);  // 8.39 MB
  _Float16* xsw = (_Float16*)(ws);
  _Float16* ssw = (_Float16*)(ws + SZ);      // later: attention output
  _Float16* qsw = (_Float16*)(ws + 2 * SZ);  // later: y1 (spans slots 2+3)
  _Float16* ksw = (_Float16*)(ws + 3 * SZ);
  _Float16* vsw = (_Float16*)(ws + 4 * SZ);
  _Float16* msw = (_Float16*)(ws + 5 * SZ);
  _Float16* atn = ssw;
  _Float16* y1  = qsw;
  // Packed f16 weights after the activation slots (~1.31 MB).
  _Float16* wqp = (_Float16*)(ws + 6 * SZ);
  _Float16* wkp = wqp + 65536;
  _Float16* wvp = wkp + 65536;
  _Float16* wmp = wvp + 65536;
  _Float16* w1p = wmp + 65536;   // 512x512 = 262144
  _Float16* w2p = w1p + 262144;  // 256x512 = 131072

  const dim3 blk(128);
  const dim3 g8(64, 8);
  const dim3 g16(64, 16);

  // Pack weights once (read 64x each afterwards).
  packw_kernel<<<dim3(8),  dim3(256), 0, stream>>>(Wq, wqp, 256, 256);
  packw_kernel<<<dim3(8),  dim3(256), 0, stream>>>(Wk, wkp, 256, 256);
  packw_kernel<<<dim3(8),  dim3(256), 0, stream>>>(Wv, wvp, 256, 256);
  packw_kernel<<<dim3(8),  dim3(256), 0, stream>>>(Wm, wmp, 256, 256);
  packw_kernel<<<dim3(32), dim3(256), 0, stream>>>(W1, w1p, 512, 512);
  packw_kernel<<<dim3(16), dim3(256), 0, stream>>>(W2, w2p, 256, 512);

  pack_kernel<<<dim3(1024), dim3(256), 0, stream>>>(x, xsw);
  pack_kernel<<<dim3(1024), dim3(256), 0, stream>>>(src, ssw);

  gemm_kernel<M_PROJ_QA, 256><<<g8, blk, 0, stream>>>(
      wqp, bq, xsw, nullptr, qsw, nullptr, nullptr, nullptr, nullptr, nullptr);
  gemm_kernel<M_PROJ_KB, 256><<<g8, blk, 0, stream>>>(
      wkp, bk, ssw, nullptr, ksw, nullptr, nullptr, nullptr, nullptr, nullptr);
  gemm_kernel<M_PROJ_VA, 256><<<g8, blk, 0, stream>>>(
      wvp, bv, ssw, nullptr, vsw, nullptr, nullptr, nullptr, nullptr, nullptr);

  attn_kernel<<<dim3(512), dim3(256), 0, stream>>>(qsw, ksw, vsw, atn);

  gemm_kernel<M_WM, 256><<<g8, blk, 0, stream>>>(
      wmp, bm, atn, nullptr, msw, nullptr, nullptr, nullptr, nullptr, nullptr);
  gemm_kernel<M_MLP1, 512><<<g16, blk, 0, stream>>>(
      w1p, b1, xsw, msw, y1, nullptr, g, be, mu, vr);
  gemm_kernel<M_MLP2, 512><<<g8, blk, 0, stream>>>(
      w2p, b2, y1, nullptr, nullptr, out, nullptr, nullptr, nullptr, nullptr);
}
